// PointNetModel_6760278524531
// MI455X (gfx1250) — compile-verified, matrix operand order checked by reading, not run on hardware
//
#include <hip/hip_runtime.h>
#include <hip/hip_bf16.h>

typedef _Float16 v16h __attribute__((ext_vector_type(16)));
typedef _Float16 v8h  __attribute__((ext_vector_type(8)));
typedef float    v8f  __attribute__((ext_vector_type(8)));

// ---------------------------------------------------------------------------
// Weight conversion: W f32 [K,N] -> Wt f16 [N,Kpad] (transposed, zero-padded K)
// so B-fragments for WMMA are contiguous-in-K 128-bit loads.
// ---------------------------------------------------------------------------
__global__ void convert_weight(const float* __restrict__ W, _Float16* __restrict__ Wt,
                               int K, int N, int Kpad) {
    int idx = blockIdx.x * blockDim.x + threadIdx.x;
    if (idx >= N * Kpad) return;
    int n = idx / Kpad, k = idx % Kpad;
    Wt[idx] = (k < K) ? (_Float16)W[(size_t)k * N + n] : (_Float16)0.f;
}

// ---------------------------------------------------------------------------
// Farthest point sampling. One block per batch. Entire cloud (SoA coords +
// running min-dist) lives in LDS (CDNA5: 320KB/WGP; N=16384 needs 256KB).
// Emits new_xyz directly (idx[0]=point 0, matching the reference scan).
// ---------------------------------------------------------------------------
__global__ __launch_bounds__(256)
void fps_kernel(const float* __restrict__ xyz, int N, int npoint,
                float* __restrict__ new_xyz) {
    extern __shared__ float sm[];
    float* sx = sm;  float* sy = sm + N;  float* sz = sm + 2 * N;  float* mind = sm + 3 * N;
    __shared__ float rmax[256];
    __shared__ int   ridx[256];
    __shared__ int   s_last;
    const int b = blockIdx.x, tid = threadIdx.x, bdim = blockDim.x;
    const float* X = xyz + (size_t)b * N * 3;
    for (int n = tid; n < N; n += bdim) {
        sx[n] = X[n * 3 + 0]; sy[n] = X[n * 3 + 1]; sz[n] = X[n * 3 + 2];
        mind[n] = 1e10f;
    }
    if (tid == 0) s_last = 0;
    __syncthreads();
    for (int i = 0; i < npoint; ++i) {
        const int last = s_last;
        const float px = sx[last], py = sy[last], pz = sz[last];
        if (tid == 0) {
            float* o = new_xyz + ((size_t)b * npoint + i) * 3;
            o[0] = px; o[1] = py; o[2] = pz;
        }
        float bv = -1.f; int bi = 0;
        for (int n = tid; n < N; n += bdim) {
            float dx = sx[n] - px, dy = sy[n] - py, dz = sz[n] - pz;
            float d = dx * dx + dy * dy + dz * dz;
            float m = fminf(mind[n], d);
            mind[n] = m;
            if (m > bv) { bv = m; bi = n; }
        }
        rmax[tid] = bv; ridx[tid] = bi;
        __syncthreads();
        for (int s = bdim >> 1; s > 0; s >>= 1) {
            if (tid < s) {
                if (rmax[tid + s] > rmax[tid] ||
                    (rmax[tid + s] == rmax[tid] && ridx[tid + s] < ridx[tid])) {
                    rmax[tid] = rmax[tid + s]; ridx[tid] = ridx[tid + s];
                }
            }
            __syncthreads();
        }
        if (tid == 0) s_last = ridx[0];
        __syncthreads();
    }
}

// ---------------------------------------------------------------------------
// Ball query + grouping. One wave32 per center. Ballot + prefix-popcount
// packs the first `nsample` in-radius indices in ascending index order
// (== reference's sorted-where), pads with the first hit, and writes the
// padded f16 GEMM input row: [dx,dy,dz, feat..., 0-pad to Kpad].
// ---------------------------------------------------------------------------
__global__ __launch_bounds__(256)
void ballquery_gather(const float* __restrict__ xyz, const float* __restrict__ new_xyz,
                      const float* __restrict__ feat, int B, int N, int npoint,
                      int Cf, int Kpad, float r2, _Float16* __restrict__ H) {
    const int lane = threadIdx.x & 31;
    const int wv = blockIdx.x * (blockDim.x >> 5) + (threadIdx.x >> 5);
    if (wv >= B * npoint) return;
    const int b = wv / npoint, j = wv % npoint;
    const float* c = new_xyz + ((size_t)b * npoint + j) * 3;
    const float cx = c[0], cy = c[1], cz = c[2];
    const float* X = xyz + (size_t)b * N * 3;
    const float* F = feat + (size_t)b * N * Cf;
    _Float16* H0 = H + (size_t)wv * 32 * Kpad;

    int cnt = 0, first = -1;
    for (int base = 0; base < N && cnt < 32; base += 32) {
        int n = base + lane;
        bool in = false;
        float dx = 0.f, dy = 0.f, dz = 0.f;
        if (n < N) {
            dx = X[n * 3 + 0] - cx; dy = X[n * 3 + 1] - cy; dz = X[n * 3 + 2] - cz;
            in = (dx * dx + dy * dy + dz * dz) < r2;
        }
        unsigned mask = (unsigned)__ballot(in ? 1 : 0);
        if (first < 0 && mask) first = base + (__ffs(mask) - 1);
        if (in) {
            int pos = cnt + __popc(mask & ((1u << lane) - 1u));
            if (pos < 32) {
                _Float16* row = H0 + (size_t)pos * Kpad;
                row[0] = (_Float16)dx; row[1] = (_Float16)dy; row[2] = (_Float16)dz;
                for (int ci = 0; ci < Cf; ++ci) row[3 + ci] = (_Float16)F[(size_t)n * Cf + ci];
                for (int ci = 3 + Cf; ci < Kpad; ++ci) row[ci] = (_Float16)0.f;
            }
        }
        cnt += __popc(mask);
    }
    if (first < 0) first = 0;
    int filled = cnt < 32 ? cnt : 32;
    const float fx = X[first * 3 + 0] - cx, fy = X[first * 3 + 1] - cy, fz = X[first * 3 + 2] - cz;
    for (int pos = filled + lane; pos < 32; pos += 32) {
        _Float16* row = H0 + (size_t)pos * Kpad;
        row[0] = (_Float16)fx; row[1] = (_Float16)fy; row[2] = (_Float16)fz;
        for (int ci = 0; ci < Cf; ++ci) row[3 + ci] = (_Float16)F[(size_t)first * Cf + ci];
        for (int ci = 3 + Cf; ci < Kpad; ++ci) row[ci] = (_Float16)0.f;
    }
}

// ---------------------------------------------------------------------------
// WMMA GEMM + bias + ReLU.  D = relu(A[M,K]f16 @ W[K,N] + b),  Wt = W^T [N,K].
// Each wave computes a 16 x (NT*16) tile: the A fragment is loaded once per
// K-step and reused across NT accumulators -> 2 + 2*NT b128 loads per NT
// WMMAs (vs 4 loads per WMMA for a single tile). K stepped by 32 via
// v_wmma_f32_16x16x32_f16. Fragments follow the ISA 16-bit layouts:
// lane=(half,r16), element e -> K = (e>>3)*16 + half*8 + (e&7).
// ---------------------------------------------------------------------------
template <int NT>
__global__ __launch_bounds__(256)
void gemm_bias_relu(const _Float16* __restrict__ A, const _Float16* __restrict__ Wt,
                    const float* __restrict__ bias, _Float16* __restrict__ Dh,
                    float* __restrict__ Df, int M, int K, int N) {
    const int lane = threadIdx.x & 31;
    const int wave = blockIdx.x * (blockDim.x >> 5) + (threadIdx.x >> 5);
    const int groupsN = (N >> 4) / NT;
    const int tiles = (M >> 4) * groupsN;
    if (wave >= tiles) return;                 // wave-uniform: EXEC stays all-ones for WMMA
    const int tm = wave / groupsN, tg = wave % groupsN;
    const int m0 = tm << 4, n0 = tg * (NT << 4);
    const int half = lane >> 4, r16 = lane & 15;
    const _Float16* arow = A + (size_t)(m0 + r16) * K + half * 8;
    const _Float16* brow[NT];
#pragma unroll
    for (int j = 0; j < NT; ++j)
        brow[j] = Wt + (size_t)(n0 + (j << 4) + r16) * K + half * 8;

    v8f acc[NT];
#pragma unroll
    for (int j = 0; j < NT; ++j) acc[j] = (v8f){};

    for (int k0 = 0; k0 < K; k0 += 32) {
        if (k0 + 32 < K) {
            __builtin_prefetch(arow + k0 + 32, 0, 0);   // global_prefetch_b8
#pragma unroll
            for (int j = 0; j < NT; ++j) __builtin_prefetch(brow[j] + k0 + 32, 0, 0);
        }
        v8h a_lo = *(const v8h*)(arow + k0);
        v8h a_hi = *(const v8h*)(arow + k0 + 16);
        v16h a = __builtin_shufflevector(a_lo, a_hi, 0,1,2,3,4,5,6,7,8,9,10,11,12,13,14,15);
#pragma unroll
        for (int j = 0; j < NT; ++j) {
            v8h b_lo = *(const v8h*)(brow[j] + k0);
            v8h b_hi = *(const v8h*)(brow[j] + k0 + 16);
            v16h bfrag = __builtin_shufflevector(b_lo, b_hi, 0,1,2,3,4,5,6,7,8,9,10,11,12,13,14,15);
            acc[j] = __builtin_amdgcn_wmma_f32_16x16x32_f16(false, a, false, bfrag,
                                                            (short)0, acc[j], false, false);
        }
    }
#pragma unroll
    for (int j = 0; j < NT; ++j) {
        const int n = n0 + (j << 4) + r16;
        const float bn = bias[n];
#pragma unroll
        for (int r = 0; r < 8; ++r) {
            int m = m0 + r + half * 8;           // C/D layout: VGPR r, lanes>=16 -> M+8
            float v = acc[j][r] + bn;
            v = v > 0.f ? v : 0.f;
            if (Dh) Dh[(size_t)m * N + n] = (_Float16)v;
            if (Df) Df[(size_t)m * N + n] = v;
        }
    }
}

// ---------------------------------------------------------------------------
// Maxpool over the nsample=32 group dimension: f16 [G*32, C] -> f32 [G, C]
// ---------------------------------------------------------------------------
__global__ __launch_bounds__(256)
void maxpool_ns32(const _Float16* __restrict__ act, float* __restrict__ out,
                  int groups, int C) {
    int idx = blockIdx.x * blockDim.x + threadIdx.x;
    if (idx >= groups * C) return;
    int g = idx / C, c = idx % C;
    const _Float16* a = act + (size_t)g * 32 * C + c;
    float v = -1e30f;
    for (int s = 0; s < 32; ++s) v = fmaxf(v, (float)a[(size_t)s * C]);
    out[idx] = v;
}

// ---------------------------------------------------------------------------
// Feature propagation interp: one wave per target point. 3 exclusion passes
// of shuffle min-reduction -> 3-NN; inverse-distance weights; writes padded
// f16 row [interp(C2) | points1(C1) | 0-pad to Kpad].
// ---------------------------------------------------------------------------
__global__ __launch_bounds__(256)
void fp_interp(const float* __restrict__ xyz1, const float* __restrict__ xyz2,
               const float* __restrict__ p2, const float* __restrict__ p1,
               int B, int N1, int N2, int C2, int C1, int Kpad,
               _Float16* __restrict__ H) {
    const int lane = threadIdx.x & 31;
    const int wv = blockIdx.x * (blockDim.x >> 5) + (threadIdx.x >> 5);
    if (wv >= B * N1) return;
    const int b = wv / N1, i = wv % N1;
    const float* t = xyz1 + ((size_t)b * N1 + i) * 3;
    const float tx = t[0], ty = t[1], tz = t[2];
    const float* X2 = xyz2 + (size_t)b * N2 * 3;

    int nn[3]; float nd[3];
    for (int tsel = 0; tsel < 3; ++tsel) {
        float bv = 1e30f; int bi = 0x7fffffff;
        for (int n = lane; n < N2; n += 32) {
            if ((tsel > 0 && n == nn[0]) || (tsel > 1 && n == nn[1])) continue;
            float dx = X2[n * 3 + 0] - tx, dy = X2[n * 3 + 1] - ty, dz = X2[n * 3 + 2] - tz;
            float d = dx * dx + dy * dy + dz * dz;
            if (d < bv || (d == bv && n < bi)) { bv = d; bi = n; }
        }
        for (int off = 16; off; off >>= 1) {
            float ov = __shfl_xor(bv, off, 32);
            int   oi = __shfl_xor(bi, off, 32);
            if (ov < bv || (ov == bv && oi < bi)) { bv = ov; bi = oi; }
        }
        nn[tsel] = bi; nd[tsel] = bv;
    }
    float w0 = 1.f / fmaxf(nd[0], 1e-10f);
    float w1 = 1.f / fmaxf(nd[1], 1e-10f);
    float w2 = 1.f / fmaxf(nd[2], 1e-10f);
    float wsum = w0 + w1 + w2;
    w0 /= wsum; w1 /= wsum; w2 /= wsum;
    const float* q0 = p2 + ((size_t)b * N2 + nn[0]) * C2;
    const float* q1 = p2 + ((size_t)b * N2 + nn[1]) * C2;
    const float* q2 = p2 + ((size_t)b * N2 + nn[2]) * C2;
    _Float16* row = H + (size_t)wv * Kpad;
    for (int c = lane; c < C2; c += 32)
        row[c] = (_Float16)(w0 * q0[c] + w1 * q1[c] + w2 * q2[c]);
    const float* s1 = p1 + ((size_t)b * N1 + i) * C1;
    for (int c = lane; c < C1; c += 32)
        row[C2 + c] = (_Float16)s1[c];
    for (int c = C2 + C1 + lane; c < Kpad; c += 32)
        row[c] = (_Float16)0.f;
}

// ---------------------------------------------------------------------------
extern "C" void kernel_launch(void* const* d_in, const int* in_sizes, int n_in,
                              void* d_out, int out_size, void* d_ws, size_t ws_size,
                              hipStream_t stream) {
    (void)in_sizes; (void)n_in; (void)out_size; (void)ws_size;
    const int B = 8;
    const float* xyz    = (const float*)d_in[0];
    const float* points = (const float*)d_in[1];

    // workspace carve
    uintptr_t base = (uintptr_t)d_ws;
    size_t off = 0;
    auto carve = [&](size_t bytes) -> void* {
        void* p = (void*)(base + off);
        off += (bytes + 255) & ~(size_t)255;
        return p;
    };
    float* l1x  = (float*)carve((size_t)B * 512 * 3 * 4);
    float* l2x  = (float*)carve((size_t)B * 128 * 3 * 4);
    float* l3x  = (float*)carve((size_t)B * 32 * 3 * 4);
    float* l4x  = (float*)carve((size_t)B * 8 * 3 * 4);
    float* l1p  = (float*)carve((size_t)B * 512 * 64 * 4);
    float* l2p  = (float*)carve((size_t)B * 128 * 128 * 4);
    float* l3p  = (float*)carve((size_t)B * 32 * 256 * 4);
    float* l4p  = (float*)carve((size_t)B * 8 * 512 * 4);
    float* l3p2 = (float*)carve((size_t)B * 32 * 256 * 4);
    float* l2p2 = (float*)carve((size_t)B * 128 * 256 * 4);
    float* l1p2 = (float*)carve((size_t)B * 512 * 128 * 4);

    // layer table: {d_in index of W, K, N, Kpad (mult of 32)}
    struct WInfo { int in_idx, K, N, Kpad; };
    const WInfo wi[21] = {
        {2, 9, 32, 32},   {4, 32, 32, 32},   {6, 32, 64, 32},      // sa1
        {8, 67, 64, 96},  {10, 64, 64, 64},  {12, 64, 128, 64},    // sa2
        {14, 131, 128, 160}, {16, 128, 128, 128}, {18, 128, 256, 128}, // sa3
        {20, 259, 256, 288}, {22, 256, 256, 256}, {24, 256, 512, 256}, // sa4
        {26, 768, 256, 768}, {28, 256, 256, 256},                  // fp1
        {30, 384, 256, 384}, {32, 256, 256, 256},                  // fp2
        {34, 320, 256, 320}, {36, 256, 128, 256},                  // fp3
        {38, 134, 128, 160}, {40, 128, 128, 128}, {42, 128, 128, 128}, // fp4
    };
    _Float16* wh[21];
    for (int i = 0; i < 21; ++i)
        wh[i] = (_Float16*)carve((size_t)wi[i].N * wi[i].Kpad * 2);

    _Float16* Hbuf = (_Float16*)carve((size_t)131072 * 160 * 2);  // max M*Kpad
    _Float16* Act0 = (_Float16*)carve((size_t)131072 * 128 * 2);  // max M*Cout

    auto cdiv = [](long a, long b) -> unsigned { return (unsigned)((a + b - 1) / b); };

    // convert + transpose + pad all weights to f16
    for (int i = 0; i < 21; ++i) {
        int tot = wi[i].N * wi[i].Kpad;
        convert_weight<<<cdiv(tot, 256), 256, 0, stream>>>(
            (const float*)d_in[wi[i].in_idx], wh[i], wi[i].K, wi[i].N, wi[i].Kpad);
    }

    auto G = [&](const _Float16* A, int w, _Float16* Dh, float* Df, int M, int K, int N) {
        const float* bptr = (const float*)d_in[wi[w].in_idx + 1];
        if (N % 64 == 0) {
            long tiles = (long)(M / 16) * (N / 64);
            gemm_bias_relu<4><<<cdiv(tiles, 8), 256, 0, stream>>>(A, wh[w], bptr, Dh, Df, M, K, N);
        } else {
            long tiles = (long)(M / 16) * (N / 32);
            gemm_bias_relu<2><<<cdiv(tiles, 8), 256, 0, stream>>>(A, wh[w], bptr, Dh, Df, M, K, N);
        }
    };

    // ---- SA1: N=16384 -> np=512, r=0.1, MLP 9->[32,32,64]
    fps_kernel<<<B, 256, (size_t)16384 * 16, stream>>>(xyz, 16384, 512, l1x);
    ballquery_gather<<<cdiv(B * 512, 8), 256, 0, stream>>>(xyz, l1x, points, B, 16384, 512, 6, 32, 0.01f, Hbuf);
    G(Hbuf, 0, Act0, nullptr, 131072, 32, 32);
    G(Act0, 1, Hbuf, nullptr, 131072, 32, 32);
    G(Hbuf, 2, Act0, nullptr, 131072, 32, 64);
    maxpool_ns32<<<cdiv((long)B * 512 * 64, 256), 256, 0, stream>>>(Act0, l1p, B * 512, 64);

    // ---- SA2: 512 -> 128, r=0.2, MLP 67->[64,64,128]
    fps_kernel<<<B, 256, (size_t)512 * 16, stream>>>(l1x, 512, 128, l2x);
    ballquery_gather<<<cdiv(B * 128, 8), 256, 0, stream>>>(l1x, l2x, l1p, B, 512, 128, 64, 96, 0.04f, Hbuf);
    G(Hbuf, 3, Act0, nullptr, 32768, 96, 64);
    G(Act0, 4, Hbuf, nullptr, 32768, 64, 64);
    G(Hbuf, 5, Act0, nullptr, 32768, 64, 128);
    maxpool_ns32<<<cdiv((long)B * 128 * 128, 256), 256, 0, stream>>>(Act0, l2p, B * 128, 128);

    // ---- SA3: 128 -> 32, r=0.4, MLP 131->[128,128,256]
    fps_kernel<<<B, 256, (size_t)128 * 16, stream>>>(l2x, 128, 32, l3x);
    ballquery_gather<<<cdiv(B * 32, 8), 256, 0, stream>>>(l2x, l3x, l2p, B, 128, 32, 128, 160, 0.16f, Hbuf);
    G(Hbuf, 6, Act0, nullptr, 8192, 160, 128);
    G(Act0, 7, Hbuf, nullptr, 8192, 128, 128);
    G(Hbuf, 8, Act0, nullptr, 8192, 128, 256);
    maxpool_ns32<<<cdiv((long)B * 32 * 256, 256), 256, 0, stream>>>(Act0, l3p, B * 32, 256);

    // ---- SA4: 32 -> 8, r=0.8, MLP 259->[256,256,512]
    fps_kernel<<<B, 256, (size_t)32 * 16, stream>>>(l3x, 32, 8, l4x);
    ballquery_gather<<<cdiv(B * 8, 8), 256, 0, stream>>>(l3x, l4x, l3p, B, 32, 8, 256, 288, 0.64f, Hbuf);
    G(Hbuf, 9, Act0, nullptr, 2048, 288, 256);
    G(Act0, 10, Hbuf, nullptr, 2048, 256, 256);
    G(Hbuf, 11, Act0, nullptr, 2048, 256, 512);
    maxpool_ns32<<<cdiv((long)B * 8 * 512, 256), 256, 0, stream>>>(Act0, l4p, B * 8, 512);

    // ---- FP1: l3(32) <- l4(8): cin 512+256=768 -> [256,256]
    fp_interp<<<cdiv(B * 32, 8), 256, 0, stream>>>(l3x, l4x, l4p, l3p, B, 32, 8, 512, 256, 768, Hbuf);
    G(Hbuf, 12, Act0, nullptr, 256, 768, 256);
    G(Act0, 13, nullptr, l3p2, 256, 256, 256);

    // ---- FP2: l2(128) <- l3(32): cin 256+128=384 -> [256,256]
    fp_interp<<<cdiv(B * 128, 8), 256, 0, stream>>>(l2x, l3x, l3p2, l2p, B, 128, 32, 256, 128, 384, Hbuf);
    G(Hbuf, 14, Act0, nullptr, 1024, 384, 256);
    G(Act0, 15, nullptr, l2p2, 1024, 256, 256);

    // ---- FP3: l1(512) <- l2(128): cin 256+64=320 -> [256,128]
    fp_interp<<<cdiv(B * 512, 8), 256, 0, stream>>>(l1x, l2x, l2p2, l1p, B, 512, 128, 256, 64, 320, Hbuf);
    G(Hbuf, 16, Act0, nullptr, 4096, 320, 256);
    G(Act0, 17, nullptr, l1p2, 4096, 256, 128);

    // ---- FP4: l0(16384) <- l1(512): cin 128+6=134 -> [128,128,128] -> d_out
    fp_interp<<<cdiv(B * 16384, 8), 256, 0, stream>>>(xyz, l1x, l1p2, points, B, 16384, 512, 128, 6, 160, Hbuf);
    G(Hbuf, 18, Act0, nullptr, 131072, 160, 128);
    G(Act0, 19, Hbuf, nullptr, 131072, 128, 128);
    G(Hbuf, 20, nullptr, (float*)d_out, 131072, 128, 128);
}
